// EnhancedDependencyParserLSTM_59210419142860
// MI455X (gfx1250) — matrix operand order repaired and checked by reading, not compiled, not run on hardware
//
#include <hip/hip_runtime.h>

// ---------------- CDNA5 WMMA types / helpers ----------------
typedef __attribute__((ext_vector_type(16))) _Float16 v16h;
typedef __attribute__((ext_vector_type(8)))  _Float16 h8;
typedef __attribute__((ext_vector_type(8)))  float    v8f;

__device__ __forceinline__ v16h dp_make_frag(h8 lo, h8 hi) {
  return __builtin_shufflevector(lo, hi, 0,1,2,3,4,5,6,7,8,9,10,11,12,13,14,15);
}

// A fragment (16x32 f16, MxK). Lane L: m=L%16; halves [k0..k0+7] and [k0+16..k0+23], k0=8*(L/16).
__device__ __forceinline__ v16h dp_load_a_frag(const _Float16* base, int ld) {
  int lane = threadIdx.x & 31;
  int m = lane & 15;
  int k0 = (lane >> 4) << 3;
  const _Float16* p = base + m * ld + k0;
  h8 lo = *(const h8*)p;
  h8 hi = *(const h8*)(p + 16);
  return dp_make_frag(lo, hi);
}

// B fragment (32x16 f16, KxN, row-major with row stride ld). Lane L: n=L%16; k [k0..k0+7] and [k0+16..k0+23], k0=8*(L/16).
__device__ __forceinline__ v16h dp_load_b_frag(const _Float16* base, int ld) {
  int lane = threadIdx.x & 31;
  int n = lane & 15;
  int kb = (lane >> 4) << 3;
  v16h r;
#pragma unroll
  for (int s = 0; s < 8; ++s) {
    r[s] = base[(kb + s) * ld + n];
    r[s + 8] = base[(16 + kb + s) * ld + n];
  }
  return r;
}

__device__ __forceinline__ float dp_sigm(float x) { return 1.0f / (1.0f + __expf(-x)); }

// ---------------- Generic batched WMMA GEMM ----------------
// C = [accum? C +] act( A(MxK,f32) * B + bias1 + bias2 )
// transB=1: B = W (N,K) row-major (computes A @ W^T). transB=0: B = W (K,N) row-major.
// Batch bz: hi=bz/inner, lo=bz%inner; offsets hi*s?hi + lo*s?lo (in elements).
__global__ void __launch_bounds__(256)
dp_gemm_wmma(const float* __restrict__ A, const float* __restrict__ Bm,
             const float* __restrict__ bias1, const float* __restrict__ bias2,
             float* __restrict__ C,
             int M, int N, int K, int lda, int ldb, int ldc,
             int transB, int act, int accum, int inner,
             long long sAhi, long long sAlo, long long sBhi, long long sBlo,
             long long sChi, long long sClo)
{
  __shared__ __align__(16) _Float16 As[2][16][32];
  __shared__ __align__(16) _Float16 Bs[4][32][16];

  int bz = blockIdx.z;
  int hi = bz / inner, lo = bz - hi * inner;
  A  += hi * sAhi + lo * sAlo;
  Bm += hi * sBhi + lo * sBlo;
  C  += hi * sChi + lo * sClo;

  int tid = threadIdx.x;
  int wid = tid >> 5;
  int wm = wid >> 2, wn = wid & 3;
  int m0 = blockIdx.y * 32;
  int n0 = blockIdx.x * 64;

  bool fullM = (m0 + 32 <= M);
  bool fullN = (n0 + 64 <= N);

  v8f acc = {};
  int nk = (K + 31) >> 5;
  for (int kc = 0; kc < nk; ++kc) {
    int k0 = kc << 5;
    bool fullK = (k0 + 32 <= K);
    __syncthreads();

    // ---- stage A (32x32) ----
    if (fullM && fullK) {
      // unguarded fast path: straight-line loads, single wait
#pragma unroll
      for (int it = 0; it < 4; ++it) {
        int idx = tid + it * 256;
        int r = idx >> 5, c = idx & 31;
        As[r >> 4][r & 15][c] = (_Float16)A[(long long)(m0 + r) * lda + (k0 + c)];
      }
    } else {
      for (int idx = tid; idx < 32 * 32; idx += 256) {
        int r = idx >> 5, c = idx & 31;
        float v = 0.f;
        int gm = m0 + r, gk = k0 + c;
        if (gm < M && gk < K) v = A[(long long)gm * lda + gk];
        As[r >> 4][r & 15][c] = (_Float16)v;
      }
    }

    // ---- stage B (32k x 64n) ----
    if (fullN && fullK) {
#pragma unroll
      for (int it = 0; it < 8; ++it) {
        int idx = tid + it * 256;
        int k = idx >> 6, n = idx & 63;
        float v = transB ? Bm[(long long)(n0 + n) * ldb + (k0 + k)]
                         : Bm[(long long)(k0 + k) * ldb + (n0 + n)];
        Bs[n >> 4][k][n & 15] = (_Float16)v;
      }
    } else {
      for (int idx = tid; idx < 32 * 64; idx += 256) {
        int k = idx >> 6, n = idx & 63;
        float v = 0.f;
        int gk = k0 + k, gn = n0 + n;
        if (gk < K && gn < N)
          v = transB ? Bm[(long long)gn * ldb + gk] : Bm[(long long)gk * ldb + gn];
        Bs[n >> 4][k][n & 15] = (_Float16)v;
      }
    }

    // prefetch next K-chunk into cache (global_prefetch_b8; speculative, dropped if invalid)
    if (kc + 1 < nk) {
      int pr = m0 + (tid & 31);
      if (pr >= M) pr = m0;
      __builtin_prefetch(&A[(long long)pr * lda + (k0 + 32)], 0, 1);
      int pn = n0 + (tid & 63);
      if (pn >= N) pn = n0;
      const float* pb = transB ? &Bm[(long long)pn * ldb + (k0 + 32)]
                               : &Bm[(long long)(k0 + 32) * ldb + pn];
      __builtin_prefetch(pb, 0, 1);
    }

    __syncthreads();
    v16h a = dp_load_a_frag(&As[wm][0][0], 32);
    v16h b = dp_load_b_frag(&Bs[wn][0][0], 16);
    acc = __builtin_amdgcn_wmma_f32_16x16x32_f16(false, a, false, b, (short)0, acc, false, false);
  }

  int lane = tid & 31;
  int n = n0 + wn * 16 + (lane & 15);
  int mbase = m0 + wm * 16 + ((lane >> 4) << 3);
  if (n < N) {
    float bb = 0.f;
    if (bias1) bb += bias1[n];
    if (bias2) bb += bias2[n];
#pragma unroll
    for (int i = 0; i < 8; ++i) {
      int m = mbase + i;
      if (m < M) {
        float v = acc[i] + bb;
        if (act == 1) v = v > 0.f ? v : 0.1f * v;
        long long ci = (long long)m * ldc + n;
        C[ci] = accum ? (C[ci] + v) : v;
      }
    }
  }
}

// ---------------- Small utility kernels ----------------
__global__ void dp_fill(float* p, long long n, float v) {
  long long i = (long long)blockIdx.x * blockDim.x + threadIdx.x;
  if (i < n) p[i] = v;
}

__global__ void dp_setcol(float* buf, int ld, int col, int rows, float v) {
  int i = blockIdx.x * blockDim.x + threadIdx.x;
  if (i < rows) buf[(long long)i * ld + col] = v;
}

// xg[w][bs][d] = char_emb[chars[bs*16+w]][d]
__global__ void dp_char_gather(const int* __restrict__ chars, const float* __restrict__ char_emb,
                               float* __restrict__ xg, int total) {
  int i = blockIdx.x * blockDim.x + threadIdx.x;
  if (i >= total) return;
  int d = i % 50;
  int rest = i / 50;
  int bs = rest % 8192;
  int w = rest / 8192;
  int ch = chars[bs * 16 + w];
  xg[i] = char_emb[ch * 50 + d];
}

// char LSTM pointwise update; gates G (8192x200) = [i f g o], hidden 50
__global__ void dp_char_point(const float* __restrict__ G, const int* __restrict__ chars,
                              float* __restrict__ h, float* __restrict__ c, int w_eff) {
  int i = blockIdx.x * blockDim.x + threadIdx.x;
  if (i >= 8192 * 50) return;
  int j = i % 50, bs = i / 50;
  if (chars[bs * 16 + w_eff] == 0) return;  // masked step: keep state
  const float* g = G + (long long)bs * 200;
  float gi = g[j], gf = g[50 + j], gg = g[100 + j], go = g[150 + j];
  float cn = dp_sigm(gf) * c[i] + dp_sigm(gi) * tanhf(gg);
  float hn = dp_sigm(go) * tanhf(cn);
  c[i] = cn;
  h[i] = hn;
}

__global__ void dp_char_concat(const float* __restrict__ hf, const float* __restrict__ hb,
                               float* __restrict__ ce, int total) {
  int i = blockIdx.x * blockDim.x + threadIdx.x;
  if (i >= total) return;
  int d = i % 100, bs = i / 100;
  ce[i] = (d < 50) ? hf[bs * 50 + d] : hb[bs * 50 + (d - 50)];
}

// x0 (t,b,350) = concat[we(100), ce(100), pe(50), phe(50), ne(50)] * mask
__global__ void dp_embed(const int* __restrict__ words, const int* __restrict__ pos,
                         const int* __restrict__ phr, const int* __restrict__ ner,
                         const int* __restrict__ lengths,
                         const float* __restrict__ we, const float* __restrict__ pe,
                         const float* __restrict__ phe, const float* __restrict__ ne,
                         const float* __restrict__ ce, float* __restrict__ x0) {
  int i = blockIdx.x * blockDim.x + threadIdx.x;
  if (i >= 8192 * 350) return;
  int d = i % 350;
  int bs = i / 350;                    // b*256 + s
  int b = bs / 256, s = bs % 256;
  float v;
  if (d < 100)      v = we[(long long)words[bs] * 100 + d];
  else if (d < 200) v = ce[(long long)bs * 100 + (d - 100)];
  else if (d < 250) v = pe[pos[bs] * 50 + (d - 200)];
  else if (d < 300) v = phe[phr[bs] * 50 + (d - 250)];
  else              v = ne[ner[bs] * 50 + (d - 300)];
  if (s >= lengths[b]) v = 0.f;
  x0[((long long)s * 32 + b) * 350 + d] = v;
}

// Whh (800,200) f32 -> Wt16[k][G*208+jp] f16, k<224, jp<208, zero padded (transposed, gate-padded)
__global__ void dp_whh16(const float* __restrict__ Whh, _Float16* __restrict__ Wt) {
  int i = blockIdx.x * blockDim.x + threadIdx.x;
  if (i >= 224 * 832) return;
  int col = i % 832, k = i / 832;
  int G = col / 208, jp = col % 208;
  float v = 0.f;
  if (k < 200 && jp < 200) v = Whh[(long long)(G * 200 + jp) * 200 + k];
  Wt[i] = (_Float16)v;
}

// (s,b,400) -> (b,s,400)
__global__ void dp_transpose(const float* __restrict__ in, float* __restrict__ outp) {
  int i = blockIdx.x * blockDim.x + threadIdx.x;
  if (i >= 8192 * 400) return;
  int d = i % 400;
  int r = i / 400;
  int b = r / 256, s = r % 256;
  outp[i] = in[((long long)s * 32 + b) * 400 + d];
}

// ---------------- Persistent word-LSTM recurrence ----------------
// One workgroup per direction (blockIdx.x). 26 waves: wave = (mtile 0..1) x (ntile 0..12).
// Each wave owns 4 gate tiles (i,f,g,o) for its (batch-tile, hidden-tile) -> pointwise in registers.
// h state f16 + c state f32 resident in LDS for all 256 steps.
__global__ void __launch_bounds__(832)
dp_word_lstm(const float* __restrict__ XgF, const float* __restrict__ XgB,
             const _Float16* __restrict__ WtF, const _Float16* __restrict__ WtB,
             const int* __restrict__ lengths, float* __restrict__ out) {
  __shared__ __align__(16) _Float16 h16[32][224];
  __shared__ float cst[32][200];

  int dir = blockIdx.x;
  const float* Xg = dir ? XgB : XgF;
  const _Float16* Wt = dir ? WtB : WtF;

  int tid = threadIdx.x;
  for (int i = tid; i < 32 * 224; i += 832) ((_Float16*)h16)[i] = (_Float16)0.f;
  for (int i = tid; i < 32 * 200; i += 832) ((float*)cst)[i] = 0.f;

  int wid = tid >> 5, lane = tid & 31;
  int mt = wid / 13, nt = wid % 13;
  int m0 = mt << 4, j0 = nt << 4;

  for (int t = 0; t < 256; ++t) {
    __syncthreads();  // h16/cst from previous step visible
    v8f a0 = {}, a1 = {}, a2 = {}, a3 = {};
#pragma unroll 1
    for (int kc = 0; kc < 7; ++kc) {
      int k0 = kc << 5;
      v16h a = dp_load_a_frag(&h16[m0][k0], 224);
      v16h b0 = dp_load_b_frag(Wt + (long long)k0 * 832 + 0 * 208 + j0, 832);
      v16h b1 = dp_load_b_frag(Wt + (long long)k0 * 832 + 1 * 208 + j0, 832);
      v16h b2 = dp_load_b_frag(Wt + (long long)k0 * 832 + 2 * 208 + j0, 832);
      v16h b3 = dp_load_b_frag(Wt + (long long)k0 * 832 + 3 * 208 + j0, 832);
      a0 = __builtin_amdgcn_wmma_f32_16x16x32_f16(false, a, false, b0, (short)0, a0, false, false);
      a1 = __builtin_amdgcn_wmma_f32_16x16x32_f16(false, a, false, b1, (short)0, a1, false, false);
      a2 = __builtin_amdgcn_wmma_f32_16x16x32_f16(false, a, false, b2, (short)0, a2, false, false);
      a3 = __builtin_amdgcn_wmma_f32_16x16x32_f16(false, a, false, b3, (short)0, a3, false, false);
    }
    __syncthreads();  // all reads of h16 done before writes
    int t_eff = dir ? (255 - t) : t;
    int j = j0 + (lane & 15);
    if (j < 200) {
      int brow = m0 + ((lane >> 4) << 3);
      const float* xg = Xg + (long long)t_eff * 32 * 800;
#pragma unroll
      for (int i = 0; i < 8; ++i) {
        int bb = brow + i;
        const float* xr = xg + (long long)bb * 800;
        float gi = a0[i] + xr[j];
        float gf = a1[i] + xr[200 + j];
        float gg = a2[i] + xr[400 + j];
        float go = a3[i] + xr[600 + j];
        float* op = out + ((long long)t_eff * 32 + bb) * 400 + dir * 200 + j;
        if (t_eff < lengths[bb]) {
          float cn = dp_sigm(gf) * cst[bb][j] + dp_sigm(gi) * tanhf(gg);
          float hn = dp_sigm(go) * tanhf(cn);
          cst[bb][j] = cn;
          h16[bb][j] = (_Float16)hn;
          *op = hn;
        } else {
          *op = 0.f;  // out = h * mask
        }
      }
    }
  }
}

// ---------------- Host orchestration ----------------
struct DPParams {
  const float *word_emb, *char_emb, *pos_emb, *phrase_emb, *ner_emb;
  const float *cf_Wih, *cf_Whh, *cf_bih, *cf_bhh;
  const float *cb_Wih, *cb_Whh, *cb_bih, *cb_bhh;
  const float *l_Wih[3][2], *l_Whh[3][2], *l_bih[3][2], *l_bhh[3][2];  // [layer][0=f,1=b]
  const float *Wad, *bad_, *Wah, *bah, *Wrd, *brd, *Wrh, *brh, *arc_W, *rel_W;
  const int *words, *chars, *pos, *phrases, *ner, *lengths;
};

extern "C" void kernel_launch(void* const* d_in, const int* in_sizes, int n_in,
                              void* d_out, int out_size, void* d_ws, size_t ws_size,
                              hipStream_t stream) {
  (void)out_size; (void)ws_size;
  auto F = [&](int i) { return (const float*)d_in[i]; };
  auto I = [&](int i) { return (const int*)d_in[i]; };

  DPParams p{};
  bool sorted_order = (n_in >= 1 && in_sizes[0] == 32 * 256 * 16);
  if (!sorted_order) {
    // insertion-order flattening
    p.word_emb = F(0); p.char_emb = F(1); p.pos_emb = F(2); p.phrase_emb = F(3); p.ner_emb = F(4);
    p.cf_Wih = F(5); p.cf_Whh = F(6); p.cf_bih = F(7); p.cf_bhh = F(8);
    p.cb_Wih = F(9); p.cb_Whh = F(10); p.cb_bih = F(11); p.cb_bhh = F(12);
    int idx = 13;
    for (int l = 0; l < 3; ++l)
      for (int d = 0; d < 2; ++d) {
        p.l_Wih[l][d] = F(idx++); p.l_Whh[l][d] = F(idx++);
        p.l_bih[l][d] = F(idx++); p.l_bhh[l][d] = F(idx++);
      }
    p.Wad = F(37); p.bad_ = F(38); p.Wah = F(39); p.bah = F(40);
    p.Wrd = F(41); p.brd = F(42); p.Wrh = F(43); p.brh = F(44);
    p.arc_W = F(45); p.rel_W = F(46);
    p.words = I(47); p.chars = I(48); p.pos = I(49); p.phrases = I(50); p.ner = I(51); p.lengths = I(52);
  } else {
    // sorted pytree flattening
    p.chars = I(0); p.lengths = I(1); p.ner = I(2);
    p.Wad = F(3); p.Wah = F(4); p.Wrd = F(5); p.Wrh = F(6); p.arc_W = F(7);
    p.bad_ = F(8); p.bah = F(9); p.brd = F(10); p.brh = F(11);
    p.cb_Whh = F(12); p.cb_Wih = F(13); p.cb_bhh = F(14); p.cb_bih = F(15);
    p.char_emb = F(16);
    p.cf_Whh = F(17); p.cf_Wih = F(18); p.cf_bhh = F(19); p.cf_bih = F(20);
    int idx = 21;
    for (int l = 0; l < 3; ++l)
      for (int d = 1; d >= 0; --d) {  // 'b' sorts before 'f'
        p.l_Whh[l][d] = F(idx++); p.l_Wih[l][d] = F(idx++);
        p.l_bhh[l][d] = F(idx++); p.l_bih[l][d] = F(idx++);
      }
    p.ner_emb = F(45); p.phrase_emb = F(46); p.pos_emb = F(47); p.rel_W = F(48); p.word_emb = F(49);
    p.phrases = I(50); p.pos = I(51); p.words = I(52);
  }

  // workspace slicing (floats, 64-element aligned)
  float* ws = (float*)d_ws;
  size_t off = 0;
  auto alloc = [&](size_t n) { float* r = ws + off; off += (n + 63) & ~(size_t)63; return r; };
  float* xgather = alloc(16LL * 8192 * 50);
  float* cgates  = alloc(8192LL * 200);
  float* chstate = alloc(4LL * 8192 * 50);
  float *hfS = chstate, *cfS = chstate + 8192 * 50, *hbS = chstate + 2 * 8192 * 50, *cbS = chstate + 3 * 8192 * 50;
  float* ce_feat = alloc(8192LL * 100);
  float* x0   = alloc(8192LL * 350);
  float* layA = alloc(8192LL * 400);
  float* layB = alloc(8192LL * 400);
  float* XgF  = alloc(8192LL * 800);
  float* XgB  = alloc(8192LL * 800);
  _Float16* WtF = (_Float16*)alloc(224 * 832 / 2);
  _Float16* WtB = (_Float16*)alloc(224 * 832 / 2);
  float* hT     = alloc(8192LL * 400);
  float* arc_d1 = alloc(8192LL * 501);
  float* arc_h  = alloc(8192LL * 500);
  float* rel_d1 = alloc(8192LL * 101);
  float* rel_h1 = alloc(8192LL * 101);
  float* T_arc  = alloc(32LL * 256 * 500);
  float* T_rel  = alloc(32LL * 256 * 101);

  auto gemm = [&](const float* A, const float* Bm, const float* b1, const float* b2, float* C,
                  int M, int N, int K, int lda, int ldb, int ldc, int transB, int act, int accum,
                  int batches, int inner,
                  long long sAhi, long long sBhi, long long sChi) {
    dim3 g((N + 63) / 64, (M + 31) / 32, batches);
    dp_gemm_wmma<<<g, 256, 0, stream>>>(A, Bm, b1, b2, C, M, N, K, lda, ldb, ldc,
                                        transB, act, accum, inner,
                                        sAhi, 0LL, sBhi, 0LL, sChi, 0LL);
  };

  // ---- char BiLSTM ----
  {
    long long n = 4LL * 8192 * 50;
    dp_fill<<<(unsigned)((n + 255) / 256), 256, 0, stream>>>(chstate, n, 0.f);
  }
  {
    int n = 16 * 8192 * 50;
    dp_char_gather<<<(n + 255) / 256, 256, 0, stream>>>(p.chars, p.char_emb, xgather, n);
  }
  for (int d = 0; d < 2; ++d) {
    const float* Wih = d ? p.cb_Wih : p.cf_Wih;
    const float* Whh = d ? p.cb_Whh : p.cf_Whh;
    const float* bih = d ? p.cb_bih : p.cf_bih;
    const float* bhh = d ? p.cb_bhh : p.cf_bhh;
    float* h = d ? hbS : hfS;
    float* c = d ? cbS : cfS;
    for (int t = 0; t < 16; ++t) {
      int w = d ? (15 - t) : t;
      gemm(xgather + (long long)w * 8192 * 50, Wih, bih, bhh, cgates,
           8192, 200, 50, 50, 50, 200, 1, 0, 0, 1, 1, 0, 0, 0);
      gemm(h, Whh, nullptr, nullptr, cgates,
           8192, 200, 50, 50, 50, 200, 1, 0, 1, 1, 1, 0, 0, 0);
      dp_char_point<<<(8192 * 50 + 255) / 256, 256, 0, stream>>>(cgates, p.chars, h, c, w);
    }
  }
  dp_char_concat<<<(8192 * 100 + 255) / 256, 256, 0, stream>>>(hfS, hbS, ce_feat, 8192 * 100);

  // ---- embeddings + concat + mask -> x0 (t,b,350) ----
  dp_embed<<<(8192 * 350 + 255) / 256, 256, 0, stream>>>(
      p.words, p.pos, p.phrases, p.ner, p.lengths,
      p.word_emb, p.pos_emb, p.phrase_emb, p.ner_emb, ce_feat, x0);

  // ---- 3 BiLSTM layers ----
  const float* cur = x0;
  float* nxt = layA;
  int din = 350;
  for (int l = 0; l < 3; ++l) {
    for (int d = 0; d < 2; ++d) {
      _Float16* Wt = d ? WtB : WtF;
      dp_whh16<<<(224 * 832 + 255) / 256, 256, 0, stream>>>(p.l_Whh[l][d], Wt);
      float* Xg = d ? XgB : XgF;
      gemm(cur, p.l_Wih[l][d], p.l_bih[l][d], p.l_bhh[l][d], Xg,
           8192, 800, din, din, din, 800, 1, 0, 0, 1, 1, 0, 0, 0);
    }
    dp_word_lstm<<<dim3(2), dim3(832), 0, stream>>>(XgF, XgB, WtF, WtB, p.lengths, nxt);
    cur = nxt;
    nxt = (nxt == layA) ? layB : layA;
    din = 400;
  }

  // ---- heads ----
  dp_transpose<<<(8192 * 400 + 255) / 256, 256, 0, stream>>>(cur, hT);
  gemm(hT, p.Wad, p.bad_, nullptr, arc_d1, 8192, 500, 400, 400, 400, 501, 1, 1, 0, 1, 1, 0, 0, 0);
  dp_setcol<<<(8192 + 255) / 256, 256, 0, stream>>>(arc_d1, 501, 500, 8192, 1.f);
  gemm(hT, p.Wah, p.bah, nullptr, arc_h, 8192, 500, 400, 400, 400, 500, 1, 1, 0, 1, 1, 0, 0, 0);
  gemm(hT, p.Wrd, p.brd, nullptr, rel_d1, 8192, 100, 400, 400, 400, 101, 1, 1, 0, 1, 1, 0, 0, 0);
  dp_setcol<<<(8192 + 255) / 256, 256, 0, stream>>>(rel_d1, 101, 100, 8192, 1.f);
  gemm(hT, p.Wrh, p.brh, nullptr, rel_h1, 8192, 100, 400, 400, 400, 101, 1, 1, 0, 1, 1, 0, 0, 0);
  dp_setcol<<<(8192 + 255) / 256, 256, 0, stream>>>(rel_h1, 101, 100, 8192, 1.f);

  float* out = (float*)d_out;
  // s_arc: T_arc[b] = arc_d1[b] @ arc_W ; s_arc[b] = T_arc[b] @ arc_h[b]^T
  gemm(arc_d1, p.arc_W, nullptr, nullptr, T_arc,
       256, 500, 501, 501, 500, 500, 0, 0, 0, 32, 1, 256LL * 501, 0, 256LL * 500);
  gemm(T_arc, arc_h, nullptr, nullptr, out,
       256, 256, 500, 500, 500, 256, 1, 0, 0, 32, 1, 256LL * 500, 256LL * 500, 256LL * 256);
  // s_rel: per relation o: T_rel[b] = rel_d1[b] @ rel_W[o] ; s_rel[b,o] = T_rel[b] @ rel_h1[b]^T
  for (int o = 0; o < 50; ++o) {
    gemm(rel_d1, p.rel_W + (long long)o * 101 * 101, nullptr, nullptr, T_rel,
         256, 101, 101, 101, 101, 101, 0, 0, 0, 32, 1, 256LL * 101, 0, 256LL * 101);
    gemm(T_rel, rel_h1, nullptr, nullptr, out + 32LL * 256 * 256 + (long long)o * 256 * 256,
         256, 256, 101, 101, 101, 256, 1, 0, 0, 32, 1, 256LL * 101, 256LL * 101, 50LL * 256 * 256);
  }
}
